// RouterMLP_26998164423124
// MI455X (gfx1250) — compile-verified, matrix operand order checked by reading, not run on hardware
//
#include <hip/hip_runtime.h>
#include <hip/hip_bf16.h>

// ---------------------------------------------------------------------------
// RouterMLP on MI455X (gfx1250):
//  - bf16 WMMA GEMMs (f32 accumulate), block tile 128x128x32, 8 waves (2x4)
//  - weights pre-converted f32->bf16 once per launch (L2-resident)
//  - tiles staged with global_load_async_to_lds_b128 (ASYNCcnt) in a
//    2-stage double-buffered pipeline: one barrier per k-step, loads for
//    tile k+1 overlap WMMAs of tile k
//  - top-1 (K=1) MoE => normalized weight is exactly 1.0 => one-hot combine,
//    implemented branch-free with LDS-cached routing indices + masked FMA
// ---------------------------------------------------------------------------

#define BDIM 256

typedef __attribute__((ext_vector_type(16))) __bf16 v16bf;
typedef __attribute__((ext_vector_type(8)))  float  v8f;

// Issue one 16-byte async global->LDS copy (per-lane). LDS byte address is
// the low 32 bits of the flat pointer (ISA 10.2).
static __device__ __forceinline__ void async_b128(void* lds, const void* g) {
  unsigned off = (unsigned)(uintptr_t)lds;
  asm volatile("global_load_async_to_lds_b128 %0, %1, off"
               :: "v"(off), "v"(g) : "memory");
}
static __device__ __forceinline__ void wait_async0() {
  asm volatile("s_wait_asynccnt 0x0" ::: "memory");
}

// Load one 16x32 bf16 fragment (A-matrix layout, ISA 7.12.2) from LDS.
static __device__ __forceinline__ v16bf frag_ld(const __bf16* p) {
  union { v16bf v; uint4 q[2]; } u;
  u.q[0] = *(const uint4*)(p);
  u.q[1] = *(const uint4*)(p + 16);
  return u.v;
}

// ---------------------------------------------------------------------------
// f32 -> bf16 bulk conversion (weights; bandwidth-trivial, runs once/launch)
// n must be a multiple of 8 (all our tensors are).
// ---------------------------------------------------------------------------
__global__ __launch_bounds__(BDIM)
void cvt_f32_bf16(const float* __restrict__ src, __bf16* __restrict__ dst, long n)
{
  long i = ((long)blockIdx.x * BDIM + threadIdx.x) * 8;
  if (i >= n) return;
  float4 f0 = *(const float4*)(src + i);
  float4 f1 = *(const float4*)(src + i + 4);
  union { __bf16 b[8]; uint4 q; } o;
  o.b[0] = (__bf16)f0.x; o.b[1] = (__bf16)f0.y;
  o.b[2] = (__bf16)f0.z; o.b[3] = (__bf16)f0.w;
  o.b[4] = (__bf16)f1.x; o.b[5] = (__bf16)f1.y;
  o.b[6] = (__bf16)f1.z; o.b[7] = (__bf16)f1.w;
  *(uint4*)(dst + i) = o.q;
}

// ---------------------------------------------------------------------------
// Generic GEMM: C[M,N](bf16) = act(A[M,K] @ W[N,K]^T + bias)
// A: bf16 (async-to-LDS) or f32 (reg prefetch + convert + packed ds_store).
// W: bf16 (async-to-LDS). Double-buffered LDS, one barrier per k-step.
// ---------------------------------------------------------------------------
template<bool A_BF16, bool RELU, bool HAS_BIAS>
__global__ __launch_bounds__(BDIM)
void gemm_wmma(const void* __restrict__ Av, const __bf16* __restrict__ W,
               const float* __restrict__ bias, __bf16* __restrict__ C,
               int M, int N, int K)
{
  constexpr int BM = 128, BN = 128, BK = 32, LDSS = 40;
  __shared__ __bf16 sA[2][BM * LDSS];
  __shared__ __bf16 sB[2][BN * LDSS];

  const int tid  = threadIdx.x;
  const int lane = tid & 31;
  const int wave = tid >> 5;
  const int wr   = wave >> 2;       // 0..1  (row group of 64)
  const int wc   = wave & 3;        // 0..3  (col group of 32)
  const int lr   = lane & 15;
  const int hi   = lane >> 4;
  const int bRow = blockIdx.y * BM;
  const int bCol = blockIdx.x * BN;

  const int ldRow = tid >> 1;       // 0..127
  const int ldCol = (tid & 1) * 16; // 0 or 16 (16 bf16 elems per thread)

  float4 aPref[4];                  // only used on the f32-A path

  auto issueB = [&](int k0, int p) {
    const __bf16* g = W + (long)(bCol + ldRow) * K + k0 + ldCol;
    __bf16* d = &sB[p][ldRow * LDSS + ldCol];
    async_b128(d, g);
    async_b128(d + 8, g + 8);
  };
  auto issueA = [&](int k0, int p) {
    if constexpr (A_BF16) {
      const __bf16* g = (const __bf16*)Av + (long)(bRow + ldRow) * K + k0 + ldCol;
      __bf16* d = &sA[p][ldRow * LDSS + ldCol];
      async_b128(d, g);
      async_b128(d + 8, g + 8);
    } else {
      const float* g = (const float*)Av + (long)(bRow + ldRow) * K + k0 + ldCol;
      aPref[0] = *(const float4*)(g);
      aPref[1] = *(const float4*)(g + 4);
      aPref[2] = *(const float4*)(g + 8);
      aPref[3] = *(const float4*)(g + 12);
    }
  };
  auto commitA = [&](int p) {        // f32 path: convert regs -> packed LDS
    if constexpr (!A_BF16) {
      __bf16* d = &sA[p][ldRow * LDSS + ldCol];
      const float* fs = (const float*)aPref;
      union { __bf16 b[16]; uint4 q[2]; } o;
      #pragma unroll
      for (int i = 0; i < 16; ++i) o.b[i] = (__bf16)fs[i];
      *(uint4*)(d)     = o.q[0];
      *(uint4*)(d + 8) = o.q[1];
    }
  };

  // prologue: stage tile 0 into buffer 0
  issueA(0, 0);
  issueB(0, 0);
  commitA(0);
  wait_async0();
  __syncthreads();

  v8f acc[4][2] = {};
  int p = 0;
  for (int k0 = 0; k0 < K; k0 += BK) {
    const int kn = k0 + BK;
    if (kn < K) { issueA(kn, p ^ 1); issueB(kn, p ^ 1); }

    v16bf af[4], bf[2];
    #pragma unroll
    for (int tm = 0; tm < 4; ++tm)
      af[tm] = frag_ld(&sA[p][(wr * 64 + tm * 16 + lr) * LDSS + hi * 8]);
    #pragma unroll
    for (int tn = 0; tn < 2; ++tn)
      bf[tn] = frag_ld(&sB[p][(wc * 32 + tn * 16 + lr) * LDSS + hi * 8]);
    #pragma unroll
    for (int tm = 0; tm < 4; ++tm)
      #pragma unroll
      for (int tn = 0; tn < 2; ++tn)
        acc[tm][tn] = __builtin_amdgcn_wmma_f32_16x16x32_bf16(
            false, af[tm], false, bf[tn], (short)0, acc[tm][tn], false, false);

    if (kn < K) { commitA(p ^ 1); wait_async0(); }
    __syncthreads();
    p ^= 1;
  }

  // epilogue: bias, optional relu, bf16 store
  #pragma unroll
  for (int tn = 0; tn < 2; ++tn) {
    const int col = bCol + wc * 32 + tn * 16 + lr;
    const float bv = HAS_BIAS ? bias[col] : 0.f;
    #pragma unroll
    for (int tm = 0; tm < 4; ++tm) {
      #pragma unroll
      for (int r = 0; r < 8; ++r) {
        const int row = bRow + wr * 64 + tm * 16 + hi * 8 + r;
        float v = acc[tm][tn][r] + bv;
        if (RELU) v = v > 0.f ? v : 0.f;
        C[(long)row * N + col] = (__bf16)v;
      }
    }
  }
}

// ---------------------------------------------------------------------------
// MoE expert stage (top-1, normalized weight == 1.0 => one-hot combine):
// comb = sum_e mask_e(row) * relu(h @ We^T). Routing indices are cached in
// LDS once per block; combine is branch-free masked FMA; single store pass.
// ---------------------------------------------------------------------------
__global__ __launch_bounds__(BDIM)
void gemm_expert_wmma(const __bf16* __restrict__ h, const __bf16* __restrict__ Wexp,
                      const int* __restrict__ topidx, __bf16* __restrict__ C,
                      int M, int H, int E)
{
  constexpr int BM = 128, BN = 128, BK = 32, LDSS = 40;
  __shared__ __bf16 sA[2][BM * LDSS];
  __shared__ __bf16 sB[2][BN * LDSS];
  __shared__ int    sTop[BM];

  const int tid  = threadIdx.x;
  const int lane = tid & 31;
  const int wave = tid >> 5;
  const int wr   = wave >> 2;
  const int wc   = wave & 3;
  const int lr   = lane & 15;
  const int hi   = lane >> 4;
  const int bRow = blockIdx.y * BM;
  const int bCol = blockIdx.x * BN;
  const int ldRow = tid >> 1;
  const int ldCol = (tid & 1) * 16;

  // cache routing indices for this row block (stable across the e-loop)
  if (tid < BM) sTop[tid] = topidx[bRow + tid];

  v8f comb[4][2] = {};               // one-hot combined result (post-relu)

  for (int e = 0; e < E; ++e) {
    const __bf16* We = Wexp + (long)e * H * H;

    auto issue = [&](int k0, int p) {
      const __bf16* ga = h + (long)(bRow + ldRow) * H + k0 + ldCol;
      __bf16* da = &sA[p][ldRow * LDSS + ldCol];
      async_b128(da, ga);
      async_b128(da + 8, ga + 8);
      const __bf16* gb = We + (long)(bCol + ldRow) * H + k0 + ldCol;
      __bf16* db = &sB[p][ldRow * LDSS + ldCol];
      async_b128(db, gb);
      async_b128(db + 8, gb + 8);
    };

    issue(0, 0);
    wait_async0();
    __syncthreads();                 // also publishes sTop (first e)

    v8f acc[4][2] = {};
    int p = 0;
    for (int k0 = 0; k0 < H; k0 += BK) {
      const int kn = k0 + BK;
      if (kn < H) issue(kn, p ^ 1);

      v16bf af[4], bf[2];
      #pragma unroll
      for (int tm = 0; tm < 4; ++tm)
        af[tm] = frag_ld(&sA[p][(wr * 64 + tm * 16 + lr) * LDSS + hi * 8]);
      #pragma unroll
      for (int tn = 0; tn < 2; ++tn)
        bf[tn] = frag_ld(&sB[p][(wc * 32 + tn * 16 + lr) * LDSS + hi * 8]);
      #pragma unroll
      for (int tm = 0; tm < 4; ++tm)
        #pragma unroll
        for (int tn = 0; tn < 2; ++tn)
          acc[tm][tn] = __builtin_amdgcn_wmma_f32_16x16x32_bf16(
              false, af[tm], false, bf[tn], (short)0, acc[tm][tn], false, false);

      if (kn < H) wait_async0();
      __syncthreads();
      p ^= 1;
    }

    // branch-free one-hot combine: comb += (sTop[row]==e) * relu(acc)
    int tsel[4][8];
    #pragma unroll
    for (int tm = 0; tm < 4; ++tm)
      #pragma unroll
      for (int r = 0; r < 8; ++r)
        tsel[tm][r] = sTop[wr * 64 + tm * 16 + hi * 8 + r];
    #pragma unroll
    for (int tm = 0; tm < 4; ++tm) {
      #pragma unroll
      for (int r = 0; r < 8; ++r) {
        const float msk = (tsel[tm][r] == e) ? 1.f : 0.f;
        #pragma unroll
        for (int tn = 0; tn < 2; ++tn) {
          float v = acc[tm][tn][r];
          v = v > 0.f ? v : 0.f;
          comb[tm][tn][r] = fmaf(msk, v, comb[tm][tn][r]);
        }
      }
    }
  }

  // single store pass (each element produced by exactly one expert)
  #pragma unroll
  for (int tn = 0; tn < 2; ++tn) {
    const int col = bCol + wc * 32 + tn * 16 + lr;
    #pragma unroll
    for (int tm = 0; tm < 4; ++tm) {
      #pragma unroll
      for (int r = 0; r < 8; ++r) {
        const int row = bRow + wr * 64 + tm * 16 + hi * 8 + r;
        C[(long)row * H + col] = (__bf16)comb[tm][tn][r];
      }
    }
  }
}

// ---------------------------------------------------------------------------
// Router gate: logits[b,e] = h[b,:]·Wg[e,:] + bg[e] (raw pre-softmax logits,
// as the reference returns) + argmax (ties -> lowest e). One wave per token.
// ---------------------------------------------------------------------------
__global__ __launch_bounds__(BDIM)
void gate_topk(const __bf16* __restrict__ h, const float* __restrict__ Wg,
               const float* __restrict__ bg, float* __restrict__ logits,
               int* __restrict__ topidx, int Btot, int H)
{
  const int gw   = (int)((blockIdx.x * (unsigned)BDIM + threadIdx.x) >> 5);
  const int lane = threadIdx.x & 31;
  if (gw >= Btot) return;
  const __bf16* hp = h + (long)gw * H;

  float a0 = 0.f, a1 = 0.f, a2 = 0.f, a3 = 0.f;
  for (int i = lane * 4; i < H; i += 128) {
    union { uint2 q; __bf16 b[4]; } u; u.q = *(const uint2*)(hp + i);
    #pragma unroll
    for (int j = 0; j < 4; ++j) {
      const float hv = (float)u.b[j];
      a0 += hv * Wg[0 * H + i + j];
      a1 += hv * Wg[1 * H + i + j];
      a2 += hv * Wg[2 * H + i + j];
      a3 += hv * Wg[3 * H + i + j];
    }
  }
  #pragma unroll
  for (int off = 16; off > 0; off >>= 1) {
    a0 += __shfl_xor(a0, off, 32);
    a1 += __shfl_xor(a1, off, 32);
    a2 += __shfl_xor(a2, off, 32);
    a3 += __shfl_xor(a3, off, 32);
  }
  if (lane == 0) {
    a0 += bg[0]; a1 += bg[1]; a2 += bg[2]; a3 += bg[3];
    float4 L; L.x = a0; L.y = a1; L.z = a2; L.w = a3;
    *(float4*)(logits + (long)gw * 4) = L;
    int best = 0; float bv = a0;
    if (a1 > bv) { bv = a1; best = 1; }
    if (a2 > bv) { bv = a2; best = 2; }
    if (a3 > bv) { bv = a3; best = 3; }
    topidx[gw] = best;
  }
}

// ---------------------------------------------------------------------------
// Output head: out[b,o] = h[b,:]·Wo[o,:] + bo[o], o<10. One wave per token.
// ---------------------------------------------------------------------------
__global__ __launch_bounds__(BDIM)
void out_head(const __bf16* __restrict__ h, const float* __restrict__ Wo,
              const float* __restrict__ bo, float* __restrict__ out,
              int Btot, int H)
{
  const int gw   = (int)((blockIdx.x * (unsigned)BDIM + threadIdx.x) >> 5);
  const int lane = threadIdx.x & 31;
  if (gw >= Btot) return;
  const __bf16* hp = h + (long)gw * H;

  float acc[10];
  #pragma unroll
  for (int o = 0; o < 10; ++o) acc[o] = 0.f;

  for (int i = lane * 4; i < H; i += 128) {
    union { uint2 q; __bf16 b[4]; } u; u.q = *(const uint2*)(hp + i);
    float hv[4];
    #pragma unroll
    for (int j = 0; j < 4; ++j) hv[j] = (float)u.b[j];
    #pragma unroll
    for (int o = 0; o < 10; ++o) {
      const float* w = Wo + (long)o * H + i;
      acc[o] += hv[0] * w[0] + hv[1] * w[1] + hv[2] * w[2] + hv[3] * w[3];
    }
  }
  #pragma unroll
  for (int o = 0; o < 10; ++o)
    #pragma unroll
    for (int off = 16; off > 0; off >>= 1) acc[o] += __shfl_xor(acc[o], off, 32);

  if (lane == 0) {
    #pragma unroll
    for (int o = 0; o < 10; ++o) out[(long)gw * 10 + o] = acc[o] + bo[o];
  }
}

// ---------------------------------------------------------------------------
extern "C" void kernel_launch(void* const* d_in, const int* in_sizes, int n_in,
                              void* d_out, int out_size, void* d_ws, size_t ws_size,
                              hipStream_t stream)
{
  const int B = 32768, H = 512, E = 4, IN = 3 * 32 * 32, OUTD = 10, R = 2;

  const float* x     = (const float*)d_in[0];
  const float* W_in  = (const float*)d_in[1];
  const float* b_in  = (const float*)d_in[2];
  const float* W_exp = (const float*)d_in[3];
  const float* W_g   = (const float*)d_in[4];
  const float* b_g   = (const float*)d_in[5];
  const float* W_t   = (const float*)d_in[6];
  const float* b_t   = (const float*)d_in[7];
  const float* W_o   = (const float*)d_in[8];
  const float* b_o   = (const float*)d_in[9];

  float* out    = (float*)d_out;               // [B,10]
  float* logit0 = out + (size_t)B * OUTD;      // [B,4]
  float* logit1 = logit0 + (size_t)B * E;      // [B,4]

  // workspace layout
  char* ws = (char*)d_ws;
  size_t off = 0;
  __bf16* h0 = (__bf16*)(ws + off); off += (size_t)B * H * sizeof(__bf16);
  __bf16* h1 = (__bf16*)(ws + off); off += (size_t)B * H * sizeof(__bf16);
  int* topidx = (int*)(ws + off);   off += (size_t)B * sizeof(int);
  __bf16* Win_bf = (__bf16*)(ws + off); off += (size_t)H * IN * sizeof(__bf16);
  __bf16* Wexp_bf = (__bf16*)(ws + off); off += (size_t)E * H * H * sizeof(__bf16);
  __bf16* Wt_bf  = (__bf16*)(ws + off); off += (size_t)H * H * sizeof(__bf16);

  const dim3 blk(BDIM);
  const dim3 gemmGrid(H / 128, B / 128);
  const int  waveBlocks = (B * 32) / BDIM;     // one wave per token

  // one-time (per launch) weight conversion: f32 -> bf16, L2-resident
  {
    long nWin = (long)H * IN, nWe = (long)E * H * H, nWt = (long)H * H;
    cvt_f32_bf16<<<(int)((nWin / 8 + BDIM - 1) / BDIM), blk, 0, stream>>>(W_in, Win_bf, nWin);
    cvt_f32_bf16<<<(int)((nWe  / 8 + BDIM - 1) / BDIM), blk, 0, stream>>>(W_exp, Wexp_bf, nWe);
    cvt_f32_bf16<<<(int)((nWt  / 8 + BDIM - 1) / BDIM), blk, 0, stream>>>(W_t, Wt_bf, nWt);
  }

  // h0 = relu(x @ W_in^T + b_in)   (A=f32 converted in-pipeline)
  gemm_wmma<false, true, true><<<gemmGrid, blk, 0, stream>>>(
      x, Win_bf, b_in, h0, B, H, IN);

  for (int r = 0; r < R; ++r) {
    float* lg = (r == 0) ? logit0 : logit1;
    gate_topk<<<waveBlocks, blk, 0, stream>>>(h0, W_g, b_g, lg, topidx, B, H);
    gemm_expert_wmma<<<gemmGrid, blk, 0, stream>>>(h0, Wexp_bf, topidx, h1, B, H, E);
    gemm_wmma<true, false, true><<<gemmGrid, blk, 0, stream>>>(
        h1, Wt_bf, b_t, h0, B, H, H);
  }

  out_head<<<waveBlocks, blk, 0, stream>>>(h0, W_o, b_o, out, B, H);
}
